// ScoreBasedContextModel_32203664785697
// MI455X (gfx1250) — matrix operand order repaired and checked by reading, not compiled, not run on hardware
//
#include <hip/hip_runtime.h>
#include <math.h>

typedef __attribute__((ext_vector_type(16))) _Float16 v16h;
typedef __attribute__((ext_vector_type(8)))  float    v8f;

#define BB 2
#define CC 128
#define HH 64
#define WW 96
#define HW (HH*WW)
#define NPIX (BB*HH*WW)
#define NBINS 4096
#define NTOT (CC*HW)          // scores per batch = 786432
#define PI_D 3.14159265358979323846

// ---------------------------------------------------------------------------
// Weight packing: OIHW f32 -> fragment-major f16 matching WMMA 16-bit A layout.
// Fragment f = ((m*9 + t)*numChunks + c): 512 halves = 32 lanes x 16 elems.
// A 16x32 f16 as packed (k = 16*sel + e): lane&15 = M; lanes 0-15: e0-15->K0-15;
//                                  lanes16-31: e0-15->K16-31.
// ---------------------------------------------------------------------------
__global__ void k_pack_weights(const float* __restrict__ w, _Float16* __restrict__ packed,
                               int cout, int cin, int numChunks)
{
  int idx = blockIdx.x * 256 + threadIdx.x;
  int total = (cout / 16) * 9 * numChunks * 512;
  if (idx >= total) return;
  int e    = idx & 15;
  int lane = (idx >> 4) & 31;
  int f    = idx >> 9;
  int c    = f % numChunks;
  int ft   = f / numChunks;
  int t    = ft % 9;
  int m    = ft / 9;
  int laneG = lane & 15;
  int sel   = lane >> 4;
  int kbase = e;                         // e0..15 -> K0..15 (pre-sel)
  int K  = kbase + sel * 16;
  int ci = c * 32 + K;
  int kh = t / 3, kw = t % 3;
  int O  = m * 16 + laneG;
  float v = 0.f;
  if (ci < cin) v = w[(((size_t)O * cin + ci) * 3 + kh) * 3 + kw];
  packed[idx] = (_Float16)v;
}

// ---------------------------------------------------------------------------
// 3x3 SAME conv, implicit GEMM via V_WMMA_F32_16X16X32_F16.
// act: NHWC f16 (cinPad channels).  Block = 256 thr (8 waves), each wave does
// one 16-cout tile over a 32-wide pixel strip (two 16x16 C tiles).
// actMode: 0=none 1=lrelu(0.2) 2=relu 3=sigmoid.  res1/res2: NHWC f16 adds.
// ---------------------------------------------------------------------------
__global__ __launch_bounds__(256)
void k_conv3x3(const _Float16* __restrict__ act, int cinPad,
               const _Float16* __restrict__ wpack, const float* __restrict__ bias, int cout,
               _Float16* __restrict__ outNHWC, float* __restrict__ outNCHW,
               const _Float16* __restrict__ res1, const _Float16* __restrict__ res2,
               int actMode)
{
  extern __shared__ char smem[];
  _Float16* lds = (_Float16*)smem;

  const int tid = threadIdx.x;
  const int strips = WW / 32;                  // 3
  const int bx = blockIdx.x;
  const int w0 = (bx % strips) * 32;
  const int h  = (bx / strips) % HH;
  const int b  = bx / (strips * HH);
  const int numChunks = cinPad >> 5;

  // Stage 3 rows x 34 cols x cinPad halo tile into LDS (channels contiguous).
  const int vecC = cinPad >> 3;                // uint4 per pixel
  const int totalVec = 3 * 34 * vecC;
  const uint4* actv = (const uint4*)act;
  uint4* ldsv = (uint4*)lds;
  for (int i = tid; i < totalVec; i += 256) {
    int v   = i % vecC;
    int pc  = i / vecC;                        // 0..101
    int row = pc / 34;
    int col = pc % 34;
    int gh = h + row - 1;
    int gw = w0 + col - 1;
    uint4 val = make_uint4(0u, 0u, 0u, 0u);
    if (gh >= 0 && gh < HH && gw >= 0 && gw < WW)
      val = actv[((size_t)((b * HH + gh) * WW + gw)) * vecC + v];
    ldsv[i] = val;
  }
  __syncthreads();

  const int wave  = tid >> 5;
  const int lane  = tid & 31;
  const int laneG = lane & 15;
  const int sel   = lane >> 4;
  const int mTile = blockIdx.y * 8 + wave;     // grid.y = cout/128 -> always valid

  v8f acc0 = {};
  v8f acc1 = {};
  const _Float16* wp = wpack + (size_t)mTile * 9 * numChunks * 512;

  for (int t = 0; t < 9; ++t) {
    const int kh = t / 3, kw = t % 3;
    for (int c = 0; c < numChunks; ++c) {
      v16h a = *(const v16h*)(wp + ((size_t)(t * numChunks + c)) * 512 + lane * 16);
      // B 32x16 f16 as loaded here (same k order as A pack): N = lane&15, lanes<16 -> K=e, lanes>=16 -> K=16+e.
      int col  = kw + laneG;
      int base = (kh * 34 + col) * cinPad + c * 32 + sel * 16;
      v16h b0 = *(const v16h*)(lds + base);
      v16h b1 = *(const v16h*)(lds + base + 16 * cinPad);   // pixels w0+16..31
      acc0 = __builtin_amdgcn_wmma_f32_16x16x32_f16(false, a, false, b0,
                                                    (short)0, acc0, false, false);
      acc1 = __builtin_amdgcn_wmma_f32_16x16x32_f16(false, a, false, b1,
                                                    (short)0, acc1, false, false);
    }
  }

  // C layout: VGPR r, lanes 0-15 -> M=r; lanes 16-31 -> M=8+r; N = lane&15.
  const int px0 = (b * HH + h) * WW + w0 + laneG;
  const int px1 = px0 + 16;
  for (int r = 0; r < 8; ++r) {
    int M  = r + sel * 8;
    int co = mTile * 16 + M;
    float bv = bias[co];
    float v0 = acc0[r] + bv;
    float v1 = acc1[r] + bv;
    if (res1) {
      v0 += (float)res1[(size_t)px0 * cout + co];
      v1 += (float)res1[(size_t)px1 * cout + co];
    }
    if (res2) {
      v0 += (float)res2[(size_t)px0 * cout + co];
      v1 += (float)res2[(size_t)px1 * cout + co];
    }
    if (actMode == 1) { v0 = v0 > 0.f ? v0 : 0.2f * v0; v1 = v1 > 0.f ? v1 : 0.2f * v1; }
    else if (actMode == 2) { v0 = fmaxf(v0, 0.f); v1 = fmaxf(v1, 0.f); }
    else if (actMode == 3) { v0 = 1.f / (1.f + expf(-v0)); v1 = 1.f / (1.f + expf(-v1)); }
    if (outNHWC) {
      outNHWC[(size_t)px0 * cout + co] = (_Float16)v0;
      outNHWC[(size_t)px1 * cout + co] = (_Float16)v1;
    }
    if (outNCHW) {
      size_t p = ((size_t)(b * cout + co) * HH + h) * WW + w0 + laneG;
      outNCHW[p]      = v0;
      outNCHW[p + 16] = v1;
    }
  }
}

// ---------------------------------------------------------------------------
// Elementwise / pack / top-k kernels
// ---------------------------------------------------------------------------
__global__ void k_zero_f(float* p, int n) { int i = blockIdx.x * 256 + threadIdx.x; if (i < n) p[i] = 0.f; }
__global__ void k_zero_i(int* p, int n)   { int i = blockIdx.x * 256 + threadIdx.x; if (i < n) p[i] = 0; }

// pa_net input: [condition(128) | ctx(128) | mask(128) | step(1) | pad] -> NHWC f16, 416 ch
__global__ void k_pack_pa(const float* __restrict__ cond, const float* __restrict__ ctx,
                          const float* __restrict__ mask, float step, _Float16* __restrict__ out)
{
  int idx = blockIdx.x * 256 + threadIdx.x;
  const int CP = 416;
  if (idx >= NPIX * CP) return;
  int cp = idx % CP;
  int p  = idx / CP;
  int b  = p / HW;
  int hw = p % HW;
  float v = 0.f;
  if      (cp < 128)  v = cond[((size_t)b * 128 + cp) * HW + hw];
  else if (cp < 256)  v = ctx [((size_t)b * 128 + (cp - 128)) * HW + hw];
  else if (cp < 384)  v = mask[((size_t)b * 128 + (cp - 256)) * HW + hw];
  else if (cp == 384) v = step;
  out[idx] = (_Float16)v;
}

// score_net input: [cond(256) | mask(128) | step(1) | pad] -> NHWC f16, 416 ch
__global__ void k_pack_score(const float* __restrict__ cond, const float* __restrict__ mask,
                             float step, _Float16* __restrict__ out)
{
  int idx = blockIdx.x * 256 + threadIdx.x;
  const int CP = 416;
  if (idx >= NPIX * CP) return;
  int cp = idx % CP;
  int p  = idx / CP;
  int b  = p / HW;
  int hw = p % HW;
  float v = 0.f;
  if      (cp < 256)  v = cond[((size_t)b * 256 + cp) * HW + hw];
  else if (cp < 384)  v = mask[((size_t)b * 128 + (cp - 256)) * HW + hw];
  else if (cp == 384) v = step;
  out[idx] = (_Float16)v;
}

// oc = m2*oc + (1-m2)*cond, m2 channel = c % 128 (uses PRE-update mask)
__global__ void k_oc(const float* __restrict__ cond, const float* __restrict__ mask,
                     float* __restrict__ oc)
{
  int idx = blockIdx.x * 256 + threadIdx.x;
  if (idx >= BB * 256 * HW) return;
  int b  = idx / (256 * HW);
  int r  = idx % (256 * HW);
  int c  = r / HW;
  int hw = r % HW;
  float m = mask[((size_t)b * 128 + (c & 127)) * HW + hw];
  oc[idx] = m * oc[idx] + (1.f - m) * cond[idx];
}

__global__ void k_hist(const float* __restrict__ score, const float* __restrict__ mask,
                       int* __restrict__ hist)
{
  int idx = blockIdx.x * 256 + threadIdx.x;
  if (idx >= BB * NTOT) return;
  if (mask[idx] == 0.f) {
    int b = idx / NTOT;
    int bin = (int)(score[idx] * (float)NBINS);
    bin = bin < 0 ? 0 : (bin > NBINS - 1 ? NBINS - 1 : bin);
    atomicAdd(&hist[b * NBINS + bin], 1);
  }
}

__global__ void k_scan(const int* __restrict__ hist, int k,
                       int* __restrict__ tbin, int* __restrict__ need, int* __restrict__ counter)
{
  int b = threadIdx.x;
  if (b >= BB) return;
  int cum = 0, T = 0, nd = 0;
  for (int bin = NBINS - 1; bin >= 0; --bin) {
    int c = hist[b * NBINS + bin];
    if (cum + c >= k) { T = bin; nd = k - cum; break; }
    cum += c;
  }
  tbin[b] = T; need[b] = nd; counter[b] = 0;
}

// select top-k (count-exact), then ctx += quant; mask = 1 at selected sites
__global__ void k_select(const float* __restrict__ score, const float* __restrict__ input,
                         const float* __restrict__ cond,
                         const int* __restrict__ tbin, const int* __restrict__ need,
                         int* __restrict__ counter,
                         float* __restrict__ ctx, float* __restrict__ mask)
{
  int idx = blockIdx.x * 256 + threadIdx.x;
  if (idx >= BB * NTOT) return;
  if (mask[idx] != 0.f) return;
  int b = idx / NTOT;
  int bin = (int)(score[idx] * (float)NBINS);
  bin = bin < 0 ? 0 : (bin > NBINS - 1 ? NBINS - 1 : bin);
  int T = tbin[b];
  bool sel = false;
  if (bin > T) sel = true;
  else if (bin == T) {
    int pos = atomicAdd(&counter[b], 1);
    sel = pos < need[b];
  }
  if (sel) {
    int r  = idx % NTOT;
    int c  = r / HW;
    int hw = r % HW;
    float mu = cond[((size_t)b * 256 + c) * HW + hw];   // cond[:, :C]
    float q  = rintf(input[idx] - mu) + mu;
    ctx[idx] += q;
    mask[idx] = 1.f;
  }
}

__global__ void k_final(const float* __restrict__ ctx, const float* __restrict__ oc,
                        float* __restrict__ out)
{
  int idx = blockIdx.x * 256 + threadIdx.x;
  if (idx >= BB * NTOT) return;
  int b  = idx / NTOT;
  int r  = idx % NTOT;
  int c  = r / HW;
  int hw = r % HW;
  float mu = oc[((size_t)b * 256 + c) * HW + hw];
  float sc = fmaxf(fabsf(oc[((size_t)b * 256 + 128 + c) * HW + hw]), 0.11f);
  float x  = ctx[idx];
  float d  = x - mu;
  const float inv_sqrt2 = 0.70710678118654752440f;
  float hi = 0.5f * (1.f + erff((d + 0.5f) / sc * inv_sqrt2));
  float lo = 0.5f * (1.f + erff((d - 0.5f) / sc * inv_sqrt2));
  out[idx] = x;
  out[(size_t)BB * NTOT + idx] = fmaxf(hi - lo, 1e-9f);
}

// ---------------------------------------------------------------------------
extern "C" void kernel_launch(void* const* d_in, const int* in_sizes, int n_in,
                              void* d_out, int out_size, void* d_ws, size_t ws_size,
                              hipStream_t stream)
{
  (void)in_sizes; (void)n_in; (void)out_size; (void)ws_size;
  const float* input     = (const float*)d_in[0];
  const float* condition = (const float*)d_in[1];
  const float* paW[3] = {(const float*)d_in[2], (const float*)d_in[4], (const float*)d_in[6]};
  const float* paB[3] = {(const float*)d_in[3], (const float*)d_in[5], (const float*)d_in[7]};
  const float* seInW  = (const float*)d_in[8];
  const float* seInB  = (const float*)d_in[9];
  const float* seRW[6]; const float* seRB[6];
  for (int j = 0; j < 6; ++j) { seRW[j] = (const float*)d_in[10 + 2 * j]; seRB[j] = (const float*)d_in[11 + 2 * j]; }
  const float* seOutW = (const float*)d_in[22];
  const float* seOutB = (const float*)d_in[23];

  char* base = (char*)d_ws;
  size_t off = 0;
  auto alloc  = [&](size_t bytes) -> void* { off = (off + 255) & ~(size_t)255; void* p = base + off; off += bytes; return p; };
  auto allocH = [&](size_t n) { return (_Float16*)alloc(n * sizeof(_Float16)); };
  auto allocF = [&](size_t n) { return (float*)alloc(n * sizeof(float)); };
  auto allocI = [&](size_t n) { return (int*)alloc(n * sizeof(int)); };

  const int CH13 = 13, CH12 = 12, CH4 = 4;   // K chunks: 416/32, 384/32, 128/32
  _Float16* wpPA0  = allocH((size_t)24 * 9 * CH13 * 512);
  _Float16* wpPA1  = allocH((size_t)24 * 9 * CH12 * 512);
  _Float16* wpPA2  = allocH((size_t)16 * 9 * CH12 * 512);
  _Float16* wpSeIn = allocH((size_t)8  * 9 * CH13 * 512);
  _Float16* wpSeR[6];
  for (int j = 0; j < 6; ++j) wpSeR[j] = allocH((size_t)8 * 9 * CH4 * 512);
  _Float16* wpSeOut = allocH((size_t)8 * 9 * CH4 * 512);

  _Float16* actPA   = allocH((size_t)NPIX * 416);
  _Float16* mid1    = allocH((size_t)NPIX * 384);
  _Float16* mid2    = allocH((size_t)NPIX * 384);
  _Float16* scoreIn = allocH((size_t)NPIX * 416);
  _Float16* h0 = allocH((size_t)NPIX * 128);
  _Float16* hA = allocH((size_t)NPIX * 128);
  _Float16* hB = allocH((size_t)NPIX * 128);
  _Float16* ht = allocH((size_t)NPIX * 128);

  float* cond  = allocF((size_t)BB * 256 * HW);
  float* score = allocF((size_t)BB * 128 * HW);
  float* ctx   = allocF((size_t)BB * 128 * HW);
  float* maskB = allocF((size_t)BB * 128 * HW);
  float* oc    = allocF((size_t)BB * 256 * HW);
  int* hist    = allocI((size_t)BB * NBINS);
  int* tbin    = allocI(BB);
  int* need    = allocI(BB);
  int* counter = allocI(BB);

  auto gridFor = [&](size_t n) { return dim3((unsigned)((n + 255) / 256)); };

  auto packw = [&](const float* w, _Float16* pk, int cout, int cin, int ch) {
    size_t tot = (size_t)(cout / 16) * 9 * ch * 512;
    k_pack_weights<<<gridFor(tot), 256, 0, stream>>>(w, pk, cout, cin, ch);
  };
  packw(paW[0], wpPA0, 384, 385, CH13);
  packw(paW[1], wpPA1, 384, 384, CH12);
  packw(paW[2], wpPA2, 256, 384, CH12);
  packw(seInW,  wpSeIn, 128, 385, CH13);
  for (int j = 0; j < 6; ++j) packw(seRW[j], wpSeR[j], 128, 128, CH4);
  packw(seOutW, wpSeOut, 128, 128, CH4);

  k_zero_f<<<gridFor((size_t)BB * 128 * HW), 256, 0, stream>>>(ctx,   BB * 128 * HW);
  k_zero_f<<<gridFor((size_t)BB * 128 * HW), 256, 0, stream>>>(maskB, BB * 128 * HW);
  k_zero_f<<<gridFor((size_t)BB * 256 * HW), 256, 0, stream>>>(oc,    BB * 256 * HW);

  auto conv = [&](const _Float16* a, int cinPad, const _Float16* wp, const float* bias, int cout,
                  _Float16* oH, float* oF, const _Float16* r1, const _Float16* r2, int mode) {
    dim3 g(BB * HH * (WW / 32), cout / 128);
    size_t ldsB = (size_t)3 * 34 * cinPad * 2;
    k_conv3x3<<<g, 256, ldsB, stream>>>(a, cinPad, wp, bias, cout, oH, oF, r1, r2, mode);
  };

  int prev = 0;
  for (int i = 1; i <= 8; ++i) {
    float step = (float)i / 8.f;
    int cur = (int)(sin((double)i * PI_D / 16.0) * (double)NTOT);
    int kk = cur - prev;
    prev = cur;

    // pa_net
    k_pack_pa<<<gridFor((size_t)NPIX * 416), 256, 0, stream>>>(condition, ctx, maskB, step, actPA);
    conv(actPA, 416, wpPA0, paB[0], 384, mid1, nullptr, nullptr, nullptr, 1);
    conv(mid1,  384, wpPA1, paB[1], 384, mid2, nullptr, nullptr, nullptr, 1);
    conv(mid2,  384, wpPA2, paB[2], 256, nullptr, cond, nullptr, nullptr, 0);

    // score_net
    k_pack_score<<<gridFor((size_t)NPIX * 416), 256, 0, stream>>>(cond, maskB, step, scoreIn);
    conv(scoreIn, 416, wpSeIn, seInB, 128, h0, nullptr, nullptr, nullptr, 0);
    const _Float16* hprev = h0;
    _Float16* hout[3] = {hA, hB, hA};
    for (int j = 0; j < 3; ++j) {
      conv(hprev, 128, wpSeR[2 * j],     seRB[2 * j],     128, ht, nullptr, nullptr, nullptr, 2);
      conv(ht,    128, wpSeR[2 * j + 1], seRB[2 * j + 1], 128, hout[j], nullptr,
           hprev, (j == 2) ? h0 : nullptr, 0);        // fuse +h (and final +h0)
      hprev = hout[j];
    }
    conv(hprev, 128, wpSeOut, seOutB, 128, nullptr, score, nullptr, nullptr, 3);

    // state updates: oc (pre-update mask), then top-k select + ctx/mask update
    k_oc<<<gridFor((size_t)BB * 256 * HW), 256, 0, stream>>>(cond, maskB, oc);
    k_zero_i<<<gridFor((size_t)BB * NBINS), 256, 0, stream>>>(hist, BB * NBINS);
    k_hist<<<gridFor((size_t)BB * NTOT), 256, 0, stream>>>(score, maskB, hist);
    k_scan<<<1, 32, 0, stream>>>(hist, kk, tbin, need, counter);
    k_select<<<gridFor((size_t)BB * NTOT), 256, 0, stream>>>(score, input, cond,
                                                             tbin, need, counter, ctx, maskB);
  }

  k_final<<<gridFor((size_t)BB * NTOT), 256, 0, stream>>>(ctx, oc, (float*)d_out);
}
